// BaseRNN_5085241279050
// MI455X (gfx1250) — matrix operand order repaired and checked by reading, not compiled, not run on hardware
//
#include <hip/hip_runtime.h>
#include <hip/hip_bf16.h>
#include <cstdint>

// Problem dims (fixed by the reference)
#define B_ 128
#define T_ 512
#define V_ 32000
#define D_ 1024
#define H_ 2048
#define SPLITK_ 4

typedef __attribute__((ext_vector_type(16))) __bf16 v16bf;
typedef __attribute__((ext_vector_type(8)))  float  v8f;

union FragAB {
    v16bf v;
    uint4 q[2];
    unsigned short u[16];
};

__device__ __forceinline__ unsigned short f2bf(float f) {
    union { float f; unsigned int u; } cv;
    cv.f = f;
    unsigned int b = cv.u;
    unsigned int r = b + 0x7FFFu + ((b >> 16) & 1u);  // round-to-nearest-even
    return (unsigned short)(r >> 16);
}

// ---------------------------------------------------------------------------
// Cast + transpose fp32 W[K,N] -> bf16 Wt[N,K]
// ---------------------------------------------------------------------------
__global__ void cast_transpose_kernel(const float* __restrict__ W,
                                      unsigned short* __restrict__ Wt,
                                      int K, int N) {
    long long idx = (long long)blockIdx.x * blockDim.x + threadIdx.x; // over N*K
    long long total = (long long)K * N;
    if (idx >= total) return;
    int n = (int)(idx / K);
    int k = (int)(idx % K);
    Wt[idx] = f2bf(W[(long long)k * N + n]);
}

__global__ void zero_h_kernel(float* __restrict__ h32,
                              unsigned short* __restrict__ hbf, int n) {
    int i = blockIdx.x * blockDim.x + threadIdx.x;
    if (i < n) { h32[i] = 0.0f; hbf[i] = 0; }
}

__global__ void copy_out_kernel(const float* __restrict__ src,
                                float* __restrict__ dst, int n) {
    int i = blockIdx.x * blockDim.x + threadIdx.x;
    if (i < n) dst[i] = src[i];
}

// ---------------------------------------------------------------------------
// Double-buffered WMMA core over K range [kbeg, kend), 64-deep B slabs
// asynchronously staged to LDS (GLOBAL_LOAD_ASYNC_TO_LDS_B128, ASYNCcnt).
// Wave tile 16x64 (4 f32 accumulators). Wt is bf16 [N,K] row-major.
// sB: two padded slabs [2][64 * SB_STRIDE] halves.
// ---------------------------------------------------------------------------
#define SB_STRIDE 72
#define SB_SLAB   (64 * SB_STRIDE)

__device__ __forceinline__ void async_stage_slab(const unsigned short* __restrict__ Wt,
                                                 unsigned short* sB, int buf,
                                                 int tile_n, long long K, long long k0,
                                                 int tid) {
#pragma unroll
    for (int j = 0; j < 2; ++j) {
        int uidx  = tid + j * 256;
        int rowN  = uidx >> 3;   // 0..63
        int chunk = uidx & 7;    // 0..7, 8 bf16 halves = 16B each
        unsigned long long ga =
            (unsigned long long)(Wt + (long long)(tile_n + rowN) * K + k0 + chunk * 8);
        unsigned lds =
            (unsigned)(unsigned long long)&sB[buf * SB_SLAB + rowN * SB_STRIDE + chunk * 8];
        // vdst VGPR supplies the LDS address; flat low 32 bits of a __shared__
        // pointer are the wave-relative LDS byte address.
        asm volatile("global_load_async_to_lds_b128 %0, %1, off"
                     :: "v"(lds), "v"(ga) : "memory");
    }
}

template <bool EMBED>
__device__ __forceinline__ void wmma_core(const float* embrow,
                                          const unsigned short* abfrow,
                                          const unsigned short* __restrict__ Wt,
                                          unsigned short* sB,
                                          int tile_n, long long K,
                                          int kbeg, int kend,
                                          v8f acc[4]) {
    const int tid  = threadIdx.x;
    const int lane = tid & 31;
    const int half = lane >> 4;
    const int l16  = lane & 15;

    const int nIter = (kend - kbeg) >> 6;  // 64-deep slabs
    async_stage_slab(Wt, sB, 0, tile_n, K, kbeg, tid);

    for (int i = 0; i < nIter; ++i) {
        const long long k0 = kbeg + ((long long)i << 6);
        const bool hasNext = (i + 1) < nIter;
        if (hasNext) {
            // previous trailing barrier guarantees buf[(i+1)&1] is consumed
            async_stage_slab(Wt, sB, (i + 1) & 1, tile_n, K, k0 + 64, tid);
            // async loads complete in order: <=2 outstanding => current slab done
            asm volatile("s_wait_asynccnt 0x2" ::: "memory");
        } else {
            asm volatile("s_wait_asynccnt 0x0" ::: "memory");
        }
        __syncthreads();

        const unsigned short* sbuf = &sB[(i & 1) * SB_SLAB];
        if (EMBED) {
            __builtin_prefetch(embrow + k0 + 64, 0, 1);
        } else {
            __builtin_prefetch(abfrow + k0 + 64, 0, 1);
        }
#pragma unroll
        for (int kk = 0; kk < 64; kk += 32) {
            const long long k = k0 + kk;
            // A fragment: 16-bit A 16x32 layout
            FragAB a;
            if (EMBED) {
                const float* p0 = embrow + k + half * 8;
                const float* p1 = embrow + k + 16 + half * 8;
#pragma unroll
                for (int j = 0; j < 8; ++j) {
                    a.u[j]     = f2bf(p0[j]);
                    a.u[8 + j] = f2bf(p1[j]);
                }
            } else {
                a.q[0] = *(const uint4*)(abfrow + k + half * 8);
                a.q[1] = *(const uint4*)(abfrow + k + 16 + half * 8);
            }
#pragma unroll
            for (int s = 0; s < 4; ++s) {
                const unsigned short* wrow =
                    &sbuf[(s * 16 + l16) * SB_STRIDE + kk + half * 8];
                FragAB b;
                b.q[0] = *(const uint4*)(wrow);
                b.q[1] = *(const uint4*)(wrow + 16);
                acc[s] = __builtin_amdgcn_wmma_f32_16x16x32_bf16(
                    false, a.v, false, b.v, (short)0, acc[s], false, false);
            }
        }
        __syncthreads();  // all waves done reading this slab before it is re-filled
    }
}

// ---------------------------------------------------------------------------
// Big projection GEMM:  C[M,N] = A[M,K] @ W[K,N] + bias
//   EMBED=true : A row r = emb[tokens[r]] (fp32, cast to bf16 in-register)
//   EMBED=false: A is bf16 [M,K] row-major
// grid = (N/64, M/128), block = 256 (8 wave32s), block tile 128x64.
// ---------------------------------------------------------------------------
template <bool EMBED>
__global__ void gemm_wmma_kernel(const void* __restrict__ Aptr,
                                 const int* __restrict__ tokens,
                                 const unsigned short* __restrict__ Wt,
                                 const float* __restrict__ bias,
                                 float* __restrict__ C,
                                 int M, int N, int K) {
    __shared__ __align__(16) unsigned short sB[2 * SB_SLAB];
    const int lane = threadIdx.x & 31;
    const int wave = threadIdx.x >> 5;
    const int half = lane >> 4;
    const int l16  = lane & 15;
    const int tile_m = blockIdx.y * 128 + wave * 16;
    const int tile_n = blockIdx.x * 64;

    const int row = tile_m + l16;
    const float* embrow = nullptr;
    const unsigned short* abfrow = nullptr;
    if (EMBED) {
        int tok = tokens[row];
        embrow = ((const float*)Aptr) + (long long)tok * K;
    } else {
        abfrow = ((const unsigned short*)Aptr) + (long long)row * K;
    }

    v8f acc[4] = {{}, {}, {}, {}};
    wmma_core<EMBED>(embrow, abfrow, Wt, sB, tile_n, K, 0, K, acc);

#pragma unroll
    for (int s = 0; s < 4; ++s) {
        int col = tile_n + s * 16 + l16;
        float bv = bias[col];
#pragma unroll
        for (int i = 0; i < 8; ++i) {
            int r = tile_m + i + half * 8;
            C[(long long)r * N + col] = acc[s][i] + bv;
        }
    }
}

// ---------------------------------------------------------------------------
// Recurrence GEMM partial: part[ks][B,H] = h_bf[B, kslab] @ Whh[kslab, N-tile]
// grid = (H/64, SPLITK), block = 256.
// ---------------------------------------------------------------------------
__global__ void rnn_gemm_partial_kernel(const unsigned short* __restrict__ h_in_bf,
                                        const unsigned short* __restrict__ WhhT,
                                        float* __restrict__ part) {
    __shared__ __align__(16) unsigned short sB[2 * SB_SLAB];
    const int lane = threadIdx.x & 31;
    const int wave = threadIdx.x >> 5;
    const int half = lane >> 4;
    const int l16  = lane & 15;
    const int tile_m = wave * 16;            // 8 waves * 16 = 128 = B
    const int tile_n = blockIdx.x * 64;
    const int ks    = blockIdx.y;
    const int kbeg  = ks * (H_ / SPLITK_);
    const int kend  = kbeg + (H_ / SPLITK_);

    const unsigned short* arow = h_in_bf + (long long)(tile_m + l16) * H_;

    v8f acc[4] = {{}, {}, {}, {}};
    wmma_core<false>(nullptr, arow, WhhT, sB, tile_n, H_, kbeg, kend, acc);

    float* pout = part + (size_t)ks * (B_ * H_);
#pragma unroll
    for (int s = 0; s < 4; ++s) {
        int col = tile_n + s * 16 + l16;
#pragma unroll
        for (int i = 0; i < 8; ++i) {
            int b = tile_m + i + half * 8;
            pout[(long long)b * H_ + col] = acc[s][i];
        }
    }
}

// ---------------------------------------------------------------------------
// Recurrence combine: h_new = mask ? tanh(xp[:,t,:] + sum_k part[k]) : h_old
// ---------------------------------------------------------------------------
__global__ void rnn_combine_kernel(const float* __restrict__ part,
                                   const float* __restrict__ xp,      // [B,T,H]
                                   const float* __restrict__ h_in_f32,
                                   const int* __restrict__ lengths,
                                   int t,
                                   float* __restrict__ h_out_f32,
                                   unsigned short* __restrict__ h_out_bf,
                                   unsigned short* __restrict__ hs_out) {
    int i = blockIdx.x * blockDim.x + threadIdx.x;
    if (i >= B_ * H_) return;
    int b   = i >> 11;       // / H_
    int col = i & (H_ - 1);  // % H_
    float pre = xp[((long long)b * T_ + t) * H_ + col];
#pragma unroll
    for (int s = 0; s < SPLITK_; ++s) pre += part[(size_t)s * (B_ * H_) + i];
    float hn = tanhf(pre);
    float hv = (t < lengths[b]) ? hn : h_in_f32[i];
    h_out_f32[i] = hv;
    unsigned short hb = f2bf(hv);
    h_out_bf[i] = hb;
    if (hs_out) hs_out[((long long)b * T_ + t) * H_ + col] = hb;
}

// ---------------------------------------------------------------------------
// Host orchestration
// ---------------------------------------------------------------------------
extern "C" void kernel_launch(void* const* d_in, const int* in_sizes, int n_in,
                              void* d_out, int out_size, void* d_ws, size_t ws_size,
                              hipStream_t stream) {
    (void)in_sizes; (void)n_in; (void)out_size; (void)ws_size;

    const int*   tokens  = (const int*)d_in[0];   // [B,T]
    const int*   lengths = (const int*)d_in[1];   // [B]
    const float* emb     = (const float*)d_in[2]; // [V,D]
    const float* W_ih0   = (const float*)d_in[3]; // [D,H]
    const float* W_hh0   = (const float*)d_in[4]; // [H,H]
    const float* b0      = (const float*)d_in[5]; // [H]
    const float* W_ih1   = (const float*)d_in[6]; // [H,H]
    const float* W_hh1   = (const float*)d_in[7]; // [H,H]
    const float* b1      = (const float*)d_in[8]; // [H]
    float* out = (float*)d_out;                   // [B,H]

    char* ws = (char*)d_ws;
    size_t off = 0;
    auto alloc = [&](size_t bytes) {
        char* p = ws + off;
        off += (bytes + 255) & ~(size_t)255;
        return p;
    };
    unsigned short* WihT0 = (unsigned short*)alloc((size_t)H_ * D_ * 2); // [H,D]
    unsigned short* WhhT0 = (unsigned short*)alloc((size_t)H_ * H_ * 2); // [H,H]
    unsigned short* WihT1 = (unsigned short*)alloc((size_t)H_ * H_ * 2);
    unsigned short* WhhT1 = (unsigned short*)alloc((size_t)H_ * H_ * 2);
    float*          xp    = (float*)alloc((size_t)B_ * T_ * H_ * 4);     // reused
    unsigned short* hs0   = (unsigned short*)alloc((size_t)B_ * T_ * H_ * 2);
    float*          part  = (float*)alloc((size_t)SPLITK_ * B_ * H_ * 4);
    float*          h32a  = (float*)alloc((size_t)B_ * H_ * 4);
    float*          h32b  = (float*)alloc((size_t)B_ * H_ * 4);
    unsigned short* hbfa  = (unsigned short*)alloc((size_t)B_ * H_ * 2);
    unsigned short* hbfb  = (unsigned short*)alloc((size_t)B_ * H_ * 2);

    // 1) Cast + transpose weights to bf16 [N,K]
    {
        long long n0 = (long long)H_ * D_;
        cast_transpose_kernel<<<(unsigned)((n0 + 255) / 256), 256, 0, stream>>>(W_ih0, WihT0, D_, H_);
        long long n1 = (long long)H_ * H_;
        unsigned g1 = (unsigned)((n1 + 255) / 256);
        cast_transpose_kernel<<<g1, 256, 0, stream>>>(W_hh0, WhhT0, H_, H_);
        cast_transpose_kernel<<<g1, 256, 0, stream>>>(W_ih1, WihT1, H_, H_);
        cast_transpose_kernel<<<g1, 256, 0, stream>>>(W_hh1, WhhT1, H_, H_);
    }

    const int M = B_ * T_;
    dim3 gemm_grid(H_ / 64, M / 128);
    dim3 rnn_grid(H_ / 64, SPLITK_);
    const unsigned cmb_grid = (B_ * H_ + 255) / 256;

    // 2) Layer 0 input projection (embedding gather fused)
    gemm_wmma_kernel<true><<<gemm_grid, 256, 0, stream>>>(
        (const void*)emb, tokens, WihT0, b0, xp, M, H_, D_);

    // 3) Layer 0 recurrence
    zero_h_kernel<<<cmb_grid, 256, 0, stream>>>(h32a, hbfa, B_ * H_);
    float* hc32 = h32a; float* hn32 = h32b;
    unsigned short* hcbf = hbfa; unsigned short* hnbf = hbfb;
    for (int t = 0; t < T_; ++t) {
        rnn_gemm_partial_kernel<<<rnn_grid, 256, 0, stream>>>(hcbf, WhhT0, part);
        rnn_combine_kernel<<<cmb_grid, 256, 0, stream>>>(
            part, xp, hc32, lengths, t, hn32, hnbf, hs0);
        { float* tf = hc32; hc32 = hn32; hn32 = tf; }
        { unsigned short* tb = hcbf; hcbf = hnbf; hnbf = tb; }
    }

    // 4) Layer 1 input projection (overwrites xp)
    gemm_wmma_kernel<false><<<gemm_grid, 256, 0, stream>>>(
        (const void*)hs0, nullptr, WihT1, b1, xp, M, H_, H_);

    // 5) Layer 1 recurrence (final hidden only)
    zero_h_kernel<<<cmb_grid, 256, 0, stream>>>(hc32, hcbf, B_ * H_);
    for (int t = 0; t < T_; ++t) {
        rnn_gemm_partial_kernel<<<rnn_grid, 256, 0, stream>>>(hcbf, WhhT1, part);
        rnn_combine_kernel<<<cmb_grid, 256, 0, stream>>>(
            part, xp, hc32, lengths, t, hn32, hnbf, nullptr);
        { float* tf = hc32; hc32 = hn32; hn32 = tf; }
        { unsigned short* tb = hcbf; hcbf = hnbf; hnbf = tb; }
    }

    // 6) Emit final hidden state (fp32)
    copy_out_kernel<<<cmb_grid, 256, 0, stream>>>(hc32, out, B_ * H_);
}
